// MultiHeadAttention_26216480374982
// MI455X (gfx1250) — matrix operand order repaired and checked by reading, not compiled, not run on hardware
//
#include <hip/hip_runtime.h>
#include <hip/hip_bf16.h>
#include <math.h>

// ---------- types ----------
typedef _Float16 h8  __attribute__((ext_vector_type(8)));
typedef _Float16 v16h __attribute__((ext_vector_type(16)));
typedef float    v8f  __attribute__((ext_vector_type(8)));
typedef float    f4   __attribute__((ext_vector_type(4)));
typedef int      i4v  __attribute__((ext_vector_type(4)));

__device__ __forceinline__ v16h combine16(h8 lo, h8 hi) {
  v16h r;
#pragma unroll
  for (int i = 0; i < 8; ++i) { r[i] = lo[i]; r[i + 8] = hi[i]; }
  return r;
}
__device__ __forceinline__ v8f zero8() {
  v8f r;
#pragma unroll
  for (int i = 0; i < 8; ++i) r[i] = 0.0f;
  return r;
}

// ---------- CDNA5 async global->LDS path (guarded; sync fallback) ----------
#if __has_builtin(__builtin_amdgcn_global_load_async_to_lds_b128)
#define HAVE_ASYNC_LDS 1
typedef __attribute__((address_space(1))) i4v* as1_i4p;
typedef __attribute__((address_space(3))) i4v* as3_i4p;
#else
#define HAVE_ASYNC_LDS 0
#endif

__device__ __forceinline__ void copyB128(_Float16* lds_dst, const _Float16* gsrc) {
#if HAVE_ASYNC_LDS
  __builtin_amdgcn_global_load_async_to_lds_b128((as1_i4p)gsrc, (as3_i4p)lds_dst, 0, 0);
#else
  *(h8*)lds_dst = *(const h8*)gsrc;
#endif
}

template <int N>
__device__ __forceinline__ void wait_async() {
#if HAVE_ASYNC_LDS
#if __has_builtin(__builtin_amdgcn_s_wait_asynccnt)
  __builtin_amdgcn_s_wait_asynccnt(N);
#else
  asm volatile("s_wait_asynccnt %0" ::"i"(N) : "memory");
#endif
#endif
}

// =====================================================================
// Kernel 1: C = A(MxK,f32) @ W(KxN,f32) + bias, fp32 out, f16 WMMA core.
// Block tile 128x128x32, 256 threads = 8 waves (2 x 4), wave tile 64x32.
// Register->LDS double buffering: load tile k+1 into VGPRs while WMMAs
// consume tile k from LDS; prefetch tile k+2 into L2.
// =====================================================================
#define BM 128
#define BN 128
#define BK 32

__global__ __launch_bounds__(256)
void gemm_bias_wmma(const float* __restrict__ A, const float* __restrict__ W,
                    const float* __restrict__ bias, float* __restrict__ C,
                    int M, int N, int K) {
  __shared__ _Float16 lds_a[2][BM * BK];   // [row][k] row-major
  __shared__ _Float16 lds_b[2][BN * BK];   // [n][k]   K-innermost (transposed)

  const int tid  = threadIdx.x;
  const int wave = tid >> 5;
  const int lane = tid & 31;
  const int half = lane >> 4;
  const int ln   = lane & 15;
  const int wm   = wave >> 2;   // 0..1
  const int wn   = wave & 3;    // 0..3
  const int m0   = blockIdx.y * BM;
  const int n0   = blockIdx.x * BN;

  // per-thread tile slices
  const int ar = tid >> 3, ac = (tid & 7) << 2;     // A: 4 rows apart by 32
  const int bk = tid >> 5, bc = (tid & 31) << 2;    // B: 4 k-rows apart by 8

  v8f acc[4][2];
#pragma unroll
  for (int i = 0; i < 4; ++i)
#pragma unroll
    for (int j = 0; j < 2; ++j) acc[i][j] = zero8();

  f4 ra[4], rb[4];
  auto load_regs = [&](int k0) {
#pragma unroll
    for (int i = 0; i < 4; ++i)
      ra[i] = *(const f4*)(A + (size_t)(m0 + ar + i * 32) * K + k0 + ac);
#pragma unroll
    for (int i = 0; i < 4; ++i)
      rb[i] = *(const f4*)(W + (size_t)(k0 + bk + i * 8) * N + n0 + bc);
    if (k0 + BK < K) {  // prefetch one more tile ahead into L2
      __builtin_prefetch(A + (size_t)(m0 + ar) * K + k0 + BK + ac, 0, 1);
      __builtin_prefetch(W + (size_t)(k0 + BK + bk) * N + n0 + bc, 0, 1);
    }
  };
  auto store_lds = [&](int buf) {
#pragma unroll
    for (int i = 0; i < 4; ++i) {
      _Float16* dst = &lds_a[buf][(ar + i * 32) * BK + ac];
      dst[0] = (_Float16)ra[i].x; dst[1] = (_Float16)ra[i].y;
      dst[2] = (_Float16)ra[i].z; dst[3] = (_Float16)ra[i].w;
    }
#pragma unroll
    for (int i = 0; i < 4; ++i) {
      int kk = bk + i * 8;
      lds_b[buf][(bc + 0) * BK + kk] = (_Float16)rb[i].x;
      lds_b[buf][(bc + 1) * BK + kk] = (_Float16)rb[i].y;
      lds_b[buf][(bc + 2) * BK + kk] = (_Float16)rb[i].z;
      lds_b[buf][(bc + 3) * BK + kk] = (_Float16)rb[i].w;
    }
  };

  load_regs(0);
  store_lds(0);
  __syncthreads();
  int cur = 0;

  for (int k0 = 0; k0 < K; k0 += BK) {
    const int kn = k0 + BK;
    if (kn < K) load_regs(kn);  // overlaps with WMMAs below

    // B frags: j<8 -> K=8*half+j ; j>=8 -> K=16+8*half+(j-8)
    v16h bf[2];
#pragma unroll
    for (int nf = 0; nf < 2; ++nf) {
      const _Float16* p = &lds_b[cur][(wn * 32 + nf * 16 + ln) * BK + half * 8];
      bf[nf] = combine16(*(const h8*)p, *(const h8*)(p + 16));
    }
    // A frags: j<8 -> K=8*half+j ; j>=8 -> K=16+8*half+(j-8)
#pragma unroll
    for (int mf = 0; mf < 4; ++mf) {
      const _Float16* p = &lds_a[cur][(wm * 64 + mf * 16 + ln) * BK + half * 8];
      v16h af = combine16(*(const h8*)p, *(const h8*)(p + 16));
#pragma unroll
      for (int nf = 0; nf < 2; ++nf)
        acc[mf][nf] = __builtin_amdgcn_wmma_f32_16x16x32_f16(
            false, af, false, bf[nf], (short)0, acc[mf][nf], false, false);
    }

    if (kn < K) {
      __syncthreads();          // everyone done reading buf[1-cur]
      store_lds(1 - cur);
      __syncthreads();          // tile kn visible to all
      cur ^= 1;
    }
  }

  // epilogue: bias add, fp32 store (C-frag: row = v + 8*half, col = ln)
#pragma unroll
  for (int nf = 0; nf < 2; ++nf) {
    int col = n0 + wn * 32 + nf * 16 + ln;
    float b = bias[col];
#pragma unroll
    for (int mf = 0; mf < 4; ++mf) {
#pragma unroll
      for (int v = 0; v < 8; ++v) {
        int row = m0 + wm * 64 + mf * 16 + v + 8 * half;
        C[(size_t)row * N + col] = acc[mf][nf][v] + b;
      }
    }
  }
}

// =====================================================================
// Kernel 2: pack qkv fp32 [B*S,3072] -> f16 Qh/Kh [bh,s,d], VhT [bh,d,s]
// Q pre-scaled by 1/sqrt(hd).
// =====================================================================
__global__ __launch_bounds__(256)
void pack_qkv(const float* __restrict__ qkv, _Float16* __restrict__ Qh,
              _Float16* __restrict__ Kh, _Float16* __restrict__ VhT) {
  const float SCALE = 0.125f;  // 64^-0.5
  size_t idx = (size_t)blockIdx.x * 256 + threadIdx.x;  // over 8192*3072
  int col  = (int)(idx % 3072);
  int rowi = (int)(idx / 3072);
  int b = rowi >> 11, s = rowi & 2047;
  int which = col >> 10;
  int h = (col >> 6) & 15;
  int d = col & 63;
  int bh = b * 16 + h;
  float v = qkv[idx];
  if (which == 0)
    Qh[((size_t)bh * 2048 + s) * 64 + d] = (_Float16)(v * SCALE);
  else if (which == 1)
    Kh[((size_t)bh * 2048 + s) * 64 + d] = (_Float16)v;
  else
    VhT[((size_t)bh * 64 + d) * 2048 + s] = (_Float16)v;
}

// =====================================================================
// Kernel 3: flash attention. Grid (S/128, B*H). 8 waves x 16 q-rows.
// K/V tiles double-buffered in LDS via async global->LDS copies
// (ASYNCcnt-tracked); online softmax in fp32; scores and PV via WMMA.
// =====================================================================
#define HD   64
#define KBLK 32

__global__ __launch_bounds__(256)
void attn_flash(const _Float16* __restrict__ Qh, const _Float16* __restrict__ Kh,
                const _Float16* __restrict__ VhT, float* __restrict__ out, int S) {
  __shared__ _Float16 lds_k[2][KBLK * HD];    // [key][d]
  __shared__ _Float16 lds_v[2][HD * KBLK];    // [d][key]
  __shared__ _Float16 lds_p[8 * 16 * KBLK];   // per-wave 16x32 P scratch

  const int tid  = threadIdx.x;
  const int wave = tid >> 5;
  const int lane = tid & 31;
  const int half = lane >> 4;
  const int ln   = lane & 15;
  const int bh   = blockIdx.y;
  const int q0   = blockIdx.x * 128 + wave * 16;
  const size_t headOff = (size_t)bh * S * HD;
  const _Float16* Vhead = VhT + (size_t)bh * HD * S;

  // per-thread staging slices for K (32x64 contiguous) and V (64 rows of 32)
  const int vd = tid >> 2, vso = (tid & 3) * 8;

  auto issue_tile = [&](int kb, int buf) {
    copyB128(&lds_k[buf][tid * 8], Kh + headOff + (size_t)kb * HD + tid * 8);
    copyB128(&lds_v[buf][vd * KBLK + vso], Vhead + (size_t)vd * S + kb + vso);
  };

  // Q A-frags (k-dim = hd = 64 -> 2 frags), pre-scaled at pack time
  v16h qf[2];
  {
    const _Float16* qrow = Qh + headOff + (size_t)(q0 + ln) * HD;
#pragma unroll
    for (int c = 0; c < 2; ++c) {
      const _Float16* p = qrow + c * 32 + half * 8;
      qf[c] = combine16(*(const h8*)p, *(const h8*)(p + 16));
    }
  }

  float mrow[8], lrow[8];
  v8f o[4];
#pragma unroll
  for (int v = 0; v < 8; ++v) { mrow[v] = -INFINITY; lrow[v] = 0.0f; }
#pragma unroll
  for (int nf = 0; nf < 4; ++nf) o[nf] = zero8();

  _Float16* pw = &lds_p[wave * 16 * KBLK];

  issue_tile(0, 0);             // prologue: tile 0 in flight
  int cur = 0;

  for (int kb = 0; kb < S; kb += KBLK) {
    __syncthreads();            // all waves done reading buf[1-cur]
    const bool more = (kb + KBLK) < S;
    if (more) issue_tile(kb + KBLK, 1 - cur);
    if (more) wait_async<2>(); else wait_async<0>();  // tile kb complete (this wave)
    __syncthreads();            // tile kb complete (all waves)

    // scores: S = Q @ K^T, two 16-key frags, k-dim 64 in 2 chunks
    v8f sc[2];
    sc[0] = zero8(); sc[1] = zero8();
#pragma unroll
    for (int sf = 0; sf < 2; ++sf) {
#pragma unroll
      for (int c = 0; c < 2; ++c) {
        const _Float16* p = &lds_k[cur][(sf * 16 + ln) * HD + c * 32 + half * 8];
        v16h bf = combine16(*(const h8*)p, *(const h8*)(p + 16));
        sc[sf] = __builtin_amdgcn_wmma_f32_16x16x32_f16(
            false, qf[c], false, bf, (short)0, sc[sf], false, false);
      }
    }

    // online softmax per row (row = v + 8*half, cols across 16-lane group)
#pragma unroll
    for (int v = 0; v < 8; ++v) {
      float t = fmaxf(sc[0][v], sc[1][v]);
#pragma unroll
      for (int off = 1; off < 16; off <<= 1) t = fmaxf(t, __shfl_xor(t, off, 32));
      float mnew = fmaxf(mrow[v], t);
      float corr = __expf(mrow[v] - mnew);
      float p0 = __expf(sc[0][v] - mnew);
      float p1 = __expf(sc[1][v] - mnew);
      float rs = p0 + p1;
#pragma unroll
      for (int off = 1; off < 16; off <<= 1) rs += __shfl_xor(rs, off, 32);
      lrow[v] = lrow[v] * corr + rs;
      mrow[v] = mnew;
#pragma unroll
      for (int nf = 0; nf < 4; ++nf) o[nf][v] *= corr;
      int row = v + 8 * half;
      pw[row * KBLK + ln]      = (_Float16)p0;
      pw[row * KBLK + 16 + ln] = (_Float16)p1;
    }

    // P (C-layout) -> A-layout via per-wave LDS (same-wave DS ops in order)
    {
      const _Float16* p = &pw[ln * KBLK + half * 8];
      v16h pf = combine16(*(const h8*)p, *(const h8*)(p + 16));
#pragma unroll
      for (int nf = 0; nf < 4; ++nf) {
        const _Float16* vp = &lds_v[cur][(nf * 16 + ln) * KBLK + half * 8];
        v16h vf = combine16(*(const h8*)vp, *(const h8*)(vp + 16));
        o[nf] = __builtin_amdgcn_wmma_f32_16x16x32_f16(
            false, pf, false, vf, (short)0, o[nf], false, false);
      }
    }
    cur ^= 1;
  }

  // normalize and store to [B*S, 1024] fp32
  const int h = bh & 15, b = bh >> 4;
#pragma unroll
  for (int nf = 0; nf < 4; ++nf) {
    int col = h * 64 + nf * 16 + ln;
#pragma unroll
    for (int v = 0; v < 8; ++v) {
      int s = q0 + v + 8 * half;
      out[((size_t)(b * S + s)) * 1024 + col] = o[nf][v] / lrow[v];
    }
  }
}

// =====================================================================
// Host-side launcher
// =====================================================================
extern "C" void kernel_launch(void* const* d_in, const int* in_sizes, int n_in,
                              void* d_out, int out_size, void* d_ws, size_t ws_size,
                              hipStream_t stream) {
  const float* x      = (const float*)d_in[0];   // [4,2048,1024]
  const float* w_qkv  = (const float*)d_in[1];   // [1024,3072]
  const float* b_qkv  = (const float*)d_in[2];   // [3072]
  const float* w_proj = (const float*)d_in[3];   // [1024,1024]
  const float* b_proj = (const float*)d_in[4];   // [1024]
  float* out = (float*)d_out;                    // [4,2048,1024]

  const int M = 8192, D = 1024, S = 2048;
  char* ws = (char*)d_ws;
  float*    qkv  = (float*)ws;                                   // 96 MB
  _Float16* Qh   = (_Float16*)(ws + (size_t)M * 3 * D * 4);      // 16 MB
  _Float16* Kh   = Qh  + (size_t)M * D;                          // 16 MB
  _Float16* VhT  = Kh  + (size_t)M * D;                          // 16 MB
  float*    attn = (float*)((char*)(VhT + (size_t)M * D));       // 32 MB

  // 1) qkv = x @ w_qkv + b_qkv
  dim3 g1(3 * D / BN, M / BM);
  gemm_bias_wmma<<<g1, 256, 0, stream>>>(x, w_qkv, b_qkv, qkv, M, 3 * D, D);

  // 2) split heads, fp32->f16, scale Q, transpose V
  pack_qkv<<<(unsigned)((size_t)M * 3 * D / 256), 256, 0, stream>>>(qkv, Qh, Kh, VhT);

  // 3) attention
  dim3 ga(S / 128, 64);
  attn_flash<<<ga, 256, 0, stream>>>(Qh, Kh, VhT, attn, S);

  // 4) out = attn @ w_proj + b_proj
  dim3 g2(D / BN, M / BM);
  gemm_bias_wmma<<<g2, 256, 0, stream>>>(attn, w_proj, b_proj, out, M, D, D);
}
